// HybridConvQCNN_65481071395366
// MI455X (gfx1250) — matrix operand order repaired and checked by reading, not compiled, not run on hardware
//
#include <hip/hip_runtime.h>
#include <hip/hip_bf16.h>

typedef __attribute__((ext_vector_type(16))) _Float16 v16h;
typedef __attribute__((ext_vector_type(8)))  float    v8f;
typedef __attribute__((ext_vector_type(8)))  unsigned u32x8;

// ds_swizzle group-of-32: xor=0x10, or=0, and=0x1f  -> swap lanes 0-15 <-> 16-31
#define SWZ_SWAPX16 0x401F

static __device__ __forceinline__ unsigned swz16u(unsigned x) {
  return (unsigned)__builtin_amdgcn_ds_swizzle((int)x, SWZ_SWAPX16);
}
static __device__ __forceinline__ float swz16f(float x) {
  return __uint_as_float(swz16u(__float_as_uint(x)));
}
static __device__ __forceinline__ unsigned pkh2(float a, float b) {
  // v_cvt_pk_rtz_f16_f32; builtin returns __fp16 ext_vector(2) -> bit-cast to u32
  return __builtin_bit_cast(unsigned, __builtin_amdgcn_cvt_pkrtz(a, b));
}
static __device__ __forceinline__ float ftanh(float x) {
#if __has_builtin(__builtin_amdgcn_tanhf)
  return __builtin_amdgcn_tanhf(x);          // native v_tanh_f32 on gfx1250
#else
  float e = __expf(2.0f * x);
  return (e - 1.0f) * __builtin_amdgcn_rcpf(e + 1.0f);
#endif
}
static __device__ __forceinline__ float fsigmoid(float x) {
  return __builtin_amdgcn_rcpf(1.0f + __expf(-x));
}
static __device__ __forceinline__ v8f wmma_f16(v16h a, v16h b, v8f c) {
  return __builtin_amdgcn_wmma_f32_16x16x32_f16(false, a, false, b, (short)0, c,
                                                false, false);
}
static __device__ __forceinline__ v16h bvec(const unsigned u[8]) {
  u32x8 t;
#pragma unroll
  for (int i = 0; i < 8; i++) t[i] = u[i];
  return __builtin_bit_cast(v16h, t);
}

// Build next layer's B operand from D of the current layer.
// D (f32, 8 VGPRs): lo lanes = feats 0..7 of sample `lane`, hi lanes = feats 8..15
// of sample `lane-16`.  B (f16): lo lanes carry K=0..7 (feats 0..7) and K=16..23,
// K=16 a constant 1.0 (bias slot); hi lanes carry K=8..15 (feats 8..15), K=24..31 = 0.
static __device__ __forceinline__ v16h nextB(v8f d, bool lo) {
  float t[8];
#pragma unroll
  for (int i = 0; i < 8; i++) t[i] = ftanh(d[i]);
  unsigned pk[4], sw[4], u[8];
#pragma unroll
  for (int j = 0; j < 4; j++) pk[j] = pkh2(t[2 * j], t[2 * j + 1]);
#pragma unroll
  for (int j = 0; j < 4; j++) sw[j] = swz16u(pk[j]);  // bring feats 8..15 down
  u[0] = pk[0];  // lo lanes: K=0..7, hi lanes: K=8..15 (own feats)
  u[1] = pk[1];
  u[2] = pk[2];
  u[3] = pk[3];
  u[4] = lo ? 0x00003C00u : 0u;  // lo lanes: K=16 -> f16 1.0 (bias), K=17 -> 0
#pragma unroll
  for (int j = 1; j < 4; j++) u[4 + j] = 0u;
  return bvec(u);
}

// A = W^T (out x in) padded to 16x32 f16 in the ISA A-layout, bias at K=16.
// W stored (in, out) row-major; element e maps to K = (e<8 ? e : e+8) + (lo?0:8).
static __device__ v16h makeA(const float* __restrict__ W,
                             const float* __restrict__ bias,
                             int kd, int nd, int lane) {
  int m = lane & 15;
  bool lo = lane < 16;
  v16h a;
#pragma unroll
  for (int i = 0; i < 16; i++) {
    int K = (i < 8 ? i : i + 8) + (lo ? 0 : 8);
    float v = 0.0f;
    if (m < nd) {
      if (K < kd)        v = W[K * nd + m];
      else if (K == 16)  v = bias[m];
    }
    a[i] = (_Float16)v;
  }
  return a;
}

// First layer collapses to rank-1: A[m][0] = sum_k W_fm[k][m], A[m][1] = b_fm[m].
static __device__ v16h makeAfm(const float* __restrict__ Wfm,
                               const float* __restrict__ bfm, int lane) {
  int m = lane & 15;
  bool lo = lane < 16;
  float s = 0.0f;
#pragma unroll
  for (int k = 0; k < 8; k++) s += Wfm[k * 16 + m];
  v16h a;
#pragma unroll
  for (int i = 0; i < 16; i++) {
    int K = (i < 8 ? i : i + 8) + (lo ? 0 : 8);
    float v = 0.0f;
    if (K == 0)      v = s;
    else if (K == 1) v = bfm[m];
    a[i] = (_Float16)v;
  }
  return a;
}

__global__ __launch_bounds__(256) void qcnn_wmma_kernel(
    const float4* __restrict__ data,
    const float* __restrict__ conv_w, const float* __restrict__ conv_b,
    const float* __restrict__ W_fm, const float* __restrict__ b_fm,
    const float* __restrict__ W_c1, const float* __restrict__ b_c1,
    const float* __restrict__ W_p1, const float* __restrict__ b_p1,
    const float* __restrict__ W_c2, const float* __restrict__ b_c2,
    const float* __restrict__ W_p2, const float* __restrict__ b_p2,
    const float* __restrict__ W_c3, const float* __restrict__ b_c3,
    const float* __restrict__ W_h, const float* __restrict__ b_h,
    float* __restrict__ out, int nSamples) {
  const int lane = threadIdx.x & 31;
  const bool lo = lane < 16;
  // Force the wave index into an SGPR so the sample loop is wave-uniform scalar
  // control flow (s_cbranch, no EXEC masking) — WMMA requires EXEC all-ones.
  const int wave =
      __builtin_amdgcn_readfirstlane((blockIdx.x * blockDim.x + threadIdx.x) >> 5);
  const int nWaves = (gridDim.x * blockDim.x) >> 5;
  const int stride = nWaves * 32;

  // Hoisted weight operands (tiny, L2-resident; amortized over the sample loop).
  const v16h a_fm = makeAfm(W_fm, b_fm, lane);
  const v16h a_c1 = makeA(W_c1, b_c1, 16, 16, lane);
  const v16h a_p1 = makeA(W_p1, b_p1, 16, 12, lane);
  const v16h a_c2 = makeA(W_c2, b_c2, 12,  8, lane);
  const v16h a_p2 = makeA(W_p2, b_p2,  8,  4, lane);
  const v16h a_c3 = makeA(W_c3, b_c3,  4,  4, lane);

  const float cw0 = conv_w[0], cw1 = conv_w[1], cw2 = conv_w[2], cw3 = conv_w[3];
  const float cb  = conv_b[0];
  const float wh0 = W_h[0], wh1 = W_h[1], wh2 = W_h[2], wh3 = W_h[3];
  const float bh  = b_h[0];

  const v8f zero8 = {};

  for (int base = wave * 32; base + 31 < nSamples; base += stride) {
    if (base + stride + 31 < nSamples)  // scalar (base is SGPR) -> s_cbranch
      __builtin_prefetch(&data[base + stride + lane], 0, 0);

    // Conv 2x2 -> single logit -> sigmoid (coalesced float4 per lane, 32 samples).
    float4 v = data[base + lane];
    float logit = fmaf(v.x, cw0, fmaf(v.y, cw1, fmaf(v.z, cw2, fmaf(v.w, cw3, cb))));
    float cv = fsigmoid(logit);

    // First-layer B operands: K=0 -> conv value, K=1 -> 1.0 (bias slot).
    unsigned u0[8] = {0u, 0u, 0u, 0u, 0u, 0u, 0u, 0u};
    unsigned u1[8] = {0u, 0u, 0u, 0u, 0u, 0u, 0u, 0u};
    float cvs = swz16f(cv);  // tile1 samples (base+16..31) into lo lanes
    u0[0] = lo ? pkh2(cv, 1.0f) : 0u;
    u1[0] = lo ? pkh2(cvs, 1.0f) : 0u;

    v8f d0 = wmma_f16(a_fm, bvec(u0), zero8);
    v8f d1 = wmma_f16(a_fm, bvec(u1), zero8);

    d0 = wmma_f16(a_c1, nextB(d0, lo), zero8);
    d1 = wmma_f16(a_c1, nextB(d1, lo), zero8);
    d0 = wmma_f16(a_p1, nextB(d0, lo), zero8);
    d1 = wmma_f16(a_p1, nextB(d1, lo), zero8);
    d0 = wmma_f16(a_c2, nextB(d0, lo), zero8);
    d1 = wmma_f16(a_c2, nextB(d1, lo), zero8);
    d0 = wmma_f16(a_p2, nextB(d0, lo), zero8);
    d1 = wmma_f16(a_p2, nextB(d1, lo), zero8);
    d0 = wmma_f16(a_c3, nextB(d0, lo), zero8);
    d1 = wmma_f16(a_c3, nextB(d1, lo), zero8);

    // Head (4->1): merge tile1 feats into hi lanes so every lane owns one sample.
    float f[4];
#pragma unroll
    for (int r = 0; r < 4; r++) {
      float x0 = ftanh(d0[r]);
      float x1 = ftanh(d1[r]);
      float x1s = swz16f(x1);
      f[r] = lo ? x0 : x1s;
    }
    float y = fmaf(f[0], wh0, fmaf(f[1], wh1, fmaf(f[2], wh2, fmaf(f[3], wh3, bh))));
    out[base + lane] = fsigmoid(y);  // coalesced 128B store per wave
  }
}

extern "C" void kernel_launch(void* const* d_in, const int* in_sizes, int n_in,
                              void* d_out, int out_size, void* d_ws, size_t ws_size,
                              hipStream_t stream) {
  const float4* data  = (const float4*)d_in[0];
  const float* conv_w = (const float*)d_in[1];
  const float* conv_b = (const float*)d_in[2];
  const float* W_fm   = (const float*)d_in[3];
  const float* b_fm   = (const float*)d_in[4];
  const float* W_c1   = (const float*)d_in[5];
  const float* b_c1   = (const float*)d_in[6];
  const float* W_p1   = (const float*)d_in[7];
  const float* b_p1   = (const float*)d_in[8];
  const float* W_c2   = (const float*)d_in[9];
  const float* b_c2   = (const float*)d_in[10];
  const float* W_p2   = (const float*)d_in[11];
  const float* b_p2   = (const float*)d_in[12];
  const float* W_c3   = (const float*)d_in[13];
  const float* b_c3   = (const float*)d_in[14];
  const float* W_h    = (const float*)d_in[15];
  const float* b_h    = (const float*)d_in[16];
  float* out = (float*)d_out;

  int nSamples = in_sizes[0] / 4;  // data is (B,1,2,2)

  dim3 grid(1024), block(256);     // 8192 waves; B=4.19M -> 16 uniform iterations
  qcnn_wmma_kernel<<<grid, block, 0, stream>>>(
      data, conv_w, conv_b, W_fm, b_fm, W_c1, b_c1, W_p1, b_p1,
      W_c2, b_c2, W_p2, b_p2, W_c3, b_c3, W_h, b_h, out, nSamples);
}
